// BasicRNNLayer_63110249447910
// MI455X (gfx1250) — matrix operand order repaired and checked by reading, not compiled, not run on hardware
//
#include <hip/hip_runtime.h>
#include <hip/hip_bf16.h>

// ---------------------------------------------------------------------------
// BasicRNNLayer for MI455X (gfx1250): bf16 WMMA GEMMs + LDS-resident scan.
// B=64, T=2048, D=H=256.
// ---------------------------------------------------------------------------

typedef unsigned short u16;
typedef __attribute__((ext_vector_type(16))) __bf16 v16bf;
typedef __attribute__((ext_vector_type(8)))  __bf16 v8bf;
typedef __attribute__((ext_vector_type(8)))  float  v8f;

union ABFrag {            // build the 16-element bf16 A fragment from two
  v16bf v;                // contiguous 8-element (16B) chunks per lane
  v8bf  half_[2];
};

#define RNN_B 64
#define RNN_T 2048
#define RNN_D 256
#define RNN_H 256
#define RNN_M (RNN_B * RNN_T)   // 131072 rows for the big GEMMs

__device__ __forceinline__ u16 f32_to_bf16(float f) {
  unsigned int u = __float_as_uint(f);
  unsigned int r = u + 0x7FFFu + ((u >> 16) & 1u);   // round-to-nearest-even
  return (u16)(r >> 16);
}

// ---------------------------------------------------------------------------
// Kernel 0: pack W_i, W_h, W_o (f32 row-major [K][N]) into bf16 transposed
// [N][K] so WMMA B-operand lane loads are contiguous 32B.
// ---------------------------------------------------------------------------
__global__ __launch_bounds__(256) void pack_weights(
    const float* __restrict__ Wi, const float* __restrict__ Wh,
    const float* __restrict__ Wo, u16* __restrict__ WT) {
  int idx = blockIdx.x * 256 + threadIdx.x;       // 0 .. 3*65536-1
  int mat = idx >> 16;
  int rem = idx & 65535;
  int n = rem >> 8, k = rem & 255;
  const float* W = (mat == 0) ? Wi : (mat == 1) ? Wh : Wo;
  WT[idx] = f32_to_bf16(W[k * 256 + n]);          // WT[mat][n][k] = W[k][n]
}

// ---------------------------------------------------------------------------
// Kernel 1: convert x (f32) -> bf16, 4 elements/thread.
// ---------------------------------------------------------------------------
__global__ __launch_bounds__(256) void convert_x(
    const float* __restrict__ x, u16* __restrict__ xb, int n4) {
  int i = blockIdx.x * 256 + threadIdx.x;
  if (i >= n4) return;
  float4 f = ((const float4*)x)[i];
  uint2 o;
  o.x = (unsigned)f32_to_bf16(f.x) | ((unsigned)f32_to_bf16(f.y) << 16);
  o.y = (unsigned)f32_to_bf16(f.z) | ((unsigned)f32_to_bf16(f.w) << 16);
  ((uint2*)xb)[i] = o;
}

// ---------------------------------------------------------------------------
// Kernel 2/4: C[M x 256] = (relu?)(A[M x 256](bf16) @ BT^T + bias), f32 out.
// BT is [N=256][K=256] bf16 (transposed weights). Workgroup = 8 waves,
// tile 128M x 128N; wave tile 64M x 32N = 4x2 WMMA tiles, K-loop 8 x 32.
// Weights (128KB) stay L2/WGP$-resident; A streams once from HBM.
// ---------------------------------------------------------------------------
__global__ __launch_bounds__(256) void gemm_bf16(
    const u16* __restrict__ A, const u16* __restrict__ BT,
    const float* __restrict__ bias, float* __restrict__ C, int relu) {
  const int lane   = threadIdx.x & 31;
  const int wv     = threadIdx.x >> 5;
  const int laneLo = lane & 15;
  const int laneHi = lane >> 4;
  const int waveM  = wv >> 2;                     // 0..1
  const int waveN  = wv & 3;                      // 0..3
  const long rowBase = (long)blockIdx.x * 128 + waveM * 64;
  const int  colBase = blockIdx.y * 128 + waveN * 32;

  v8f acc[4][2];
  float bv[2];
#pragma unroll
  for (int j = 0; j < 2; ++j) bv[j] = bias[colBase + j * 16 + laneLo];
#pragma unroll
  for (int i = 0; i < 4; ++i)
#pragma unroll
    for (int j = 0; j < 2; ++j)
#pragma unroll
      for (int r = 0; r < 8; ++r) acc[i][j][r] = bv[j];  // bias folded into C

  for (int kt = 0; kt < 8; ++kt) {
    const int kA = kt * 32 + laneHi * 16;
    ABFrag a[4];
#pragma unroll
    for (int i = 0; i < 4; ++i) {
      const u16* p = A + (rowBase + i * 16 + laneLo) * 256 + kA;
      a[i].half_[0] = *(const v8bf*)p;            // K = kA .. kA+7
      a[i].half_[1] = *(const v8bf*)(p + 8);      // K = kA+8 .. kA+15
    }
    const int kB = kt * 32 + laneHi * 16;
    v16bf b[2];
#pragma unroll
    for (int j = 0; j < 2; ++j)
      b[j] = *(const v16bf*)(BT + (long)(colBase + j * 16 + laneLo) * 256 + kB);
#pragma unroll
    for (int i = 0; i < 4; ++i)
#pragma unroll
      for (int j = 0; j < 2; ++j)
        acc[i][j] = __builtin_amdgcn_wmma_f32_16x16x32_bf16(
            false, a[i].v, false, b[j], (short)0, acc[i][j], false, false);
  }

#pragma unroll
  for (int i = 0; i < 4; ++i)
#pragma unroll
    for (int j = 0; j < 2; ++j)
#pragma unroll
      for (int r = 0; r < 8; ++r) {
        long row = rowBase + i * 16 + r + laneHi * 8;
        int  col = colBase + j * 16 + laneLo;
        float v = acc[i][j][r];
        if (relu) v = fmaxf(v, 0.f);
        C[row * 256 + col] = v;
      }
}

// ---------------------------------------------------------------------------
// Kernel 3: the serial scan. 4 workgroups (16 batch rows each, independent),
// 512 threads = 16 waves; wave w owns N-columns [16w,16w+16). W_h^T (128KB
// bf16) lives in LDS for all 2048 steps; h is a double-buffered bf16 16x256
// LDS tile so one workgroup barrier per step suffices.
//
// Latency engineering (scan dominates total runtime):
//  - Two independent WMMA accumulator chains (K split 0..127 / 128..255)
//    halve the RAW dependency depth per step (8 -> 4).
//  - xi_t loads are software-pipelined one step ahead in registers so the
//    HBM latency hides under the previous step's WMMA chain instead of
//    stalling after each barrier.
// ---------------------------------------------------------------------------
__global__ __launch_bounds__(512) void rnn_scan(
    const float* __restrict__ xi, const u16* __restrict__ WhT,
    u16* __restrict__ states) {
  extern __shared__ char smem[];
  u16* w  = (u16*)smem;                     // [256][256] bf16 = 128KB
  u16* hb = (u16*)(smem + 256 * 256 * 2);   // 2 x [16][256] bf16 = 16KB

  const int tid    = threadIdx.x;
  const int lane   = tid & 31;
  const int wv     = tid >> 5;              // 0..15 -> N tile
  const int laneLo = lane & 15;
  const int laneHi = lane >> 4;
  const int g      = blockIdx.x;            // batch group of 16 rows

  // Stage W_h^T into LDS (8192 x 16B) and zero h buffer 0.
  {
    const uint4* src = (const uint4*)WhT;
    uint4* dst = (uint4*)w;
    for (int i = tid; i < 8192; i += 512) dst[i] = src[i];
    uint4 z; z.x = z.y = z.z = z.w = 0u;
    uint4* h0 = (uint4*)hb;
    for (int i = tid; i < 512; i += 512) h0[i] = z;
  }
  __syncthreads();

  const int n = wv * 16 + laneLo;           // this lane's output column

  // Per-lane xi/states row bases: row(r) = (g*16 + r + 8*laneHi)*T + t.
  long rbase[8];
#pragma unroll
  for (int r = 0; r < 8; ++r)
    rbase[r] = ((long)(g * 16 + r + laneHi * 8) * RNN_T) * 256 + n;

  // Prefetch xi for t = 0.
  float xv[8];
#pragma unroll
  for (int r = 0; r < 8; ++r) xv[r] = xi[rbase[r]];

  int cur = 0;
  for (int t = 0; t < RNN_T; ++t) {
    // Issue next step's xi loads first; they complete under the WMMA chain.
    float xn[8];
    if (t + 1 < RNN_T) {
#pragma unroll
      for (int r = 0; r < 8; ++r) xn[r] = xi[rbase[r] + (long)(t + 1) * 256];
    }

    const u16* h = hb + cur * (16 * 256);
    v8f acc0, acc1;                          // two independent K-chains
#pragma unroll
    for (int r = 0; r < 8; ++r) { acc0[r] = 0.f; acc1[r] = 0.f; }

#pragma unroll
    for (int kt = 0; kt < 4; ++kt) {
      // chain 0: K = kt*32 .. +32
      {
        const int kA = kt * 32 + laneHi * 16;
        ABFrag a;
        a.half_[0] = *(const v8bf*)&h[laneLo * 256 + kA];      // ds_load_b128
        a.half_[1] = *(const v8bf*)&h[laneLo * 256 + kA + 8];
        const int kB = kt * 32 + laneHi * 16;
        v16bf b = *(const v16bf*)&w[n * 256 + kB];
        acc0 = __builtin_amdgcn_wmma_f32_16x16x32_bf16(
            false, a.v, false, b, (short)0, acc0, false, false);
      }
      // chain 1: K = 128 + kt*32 .. +32
      {
        const int kA = 128 + kt * 32 + laneHi * 16;
        ABFrag a;
        a.half_[0] = *(const v8bf*)&h[laneLo * 256 + kA];
        a.half_[1] = *(const v8bf*)&h[laneLo * 256 + kA + 8];
        const int kB = 128 + kt * 32 + laneHi * 16;
        v16bf b = *(const v16bf*)&w[n * 256 + kB];
        acc1 = __builtin_amdgcn_wmma_f32_16x16x32_bf16(
            false, a.v, false, b, (short)0, acc1, false, false);
      }
    }

    u16* hn = hb + (cur ^ 1) * (16 * 256);
#pragma unroll
    for (int r = 0; r < 8; ++r) {
      int m = r + laneHi * 8;                       // local batch row 0..15
      float v = acc0[r] + acc1[r] + xv[r];          // xi already has b_h
      v = fmaxf(v, 0.f);
      u16 bvv = f32_to_bf16(v);
      hn[m * 256 + n] = bvv;                        // feed next step (LDS)
      states[rbase[r] + (long)t * 256] = bvv;       // feed output GEMM
    }
#pragma unroll
    for (int r = 0; r < 8; ++r) xv[r] = xn[r];
    cur ^= 1;
    __syncthreads();
  }
}

// ---------------------------------------------------------------------------
// Workspace layout (bytes):
//   [0,       393216)  : WiT | WhT | WoT  (3 x 256*256 bf16)
//   [393216,  67502080): x_bf16           (131072*256 bf16)
//   [67502080,201719808): xi f32          (131072*256 f32)
//   [201719808,268828672): states bf16    (131072*256 bf16)
// ---------------------------------------------------------------------------
extern "C" void kernel_launch(void* const* d_in, const int* in_sizes, int n_in,
                              void* d_out, int out_size, void* d_ws, size_t ws_size,
                              hipStream_t stream) {
  const float* x  = (const float*)d_in[0];
  const float* Wh = (const float*)d_in[1];
  const float* Wi = (const float*)d_in[2];
  const float* Wo = (const float*)d_in[3];
  const float* bh = (const float*)d_in[4];
  const float* bo = (const float*)d_in[5];
  float* out = (float*)d_out;

  char* ws = (char*)d_ws;
  u16*   WT  = (u16*)ws;
  u16*   WiT = WT;
  u16*   WhT = WT + 65536;
  u16*   WoT = WT + 131072;
  u16*   xb  = (u16*)(ws + 393216);
  float* xi  = (float*)(ws + 67502080);
  u16*   st  = (u16*)(ws + 201719808);

  pack_weights<<<768, 256, 0, stream>>>(Wi, Wh, Wo, WT);
  convert_x<<<32768, 256, 0, stream>>>(x, xb, RNN_M * RNN_D / 4);

  dim3 gGemm(RNN_M / 128, RNN_H / 128);
  gemm_bf16<<<gGemm, 256, 0, stream>>>(xb, WiT, bh, xi, 0);

  rnn_scan<<<RNN_B / 16, 512, 256 * 256 * 2 + 2 * 16 * 256 * 2, stream>>>(
      xi, WhT, st);

  gemm_bf16<<<gGemm, 256, 0, stream>>>(st, WoT, bo, out, 1);
}
